// MultiHeadAttention_33045478375951
// MI455X (gfx1250) — matrix operand order repaired and checked by reading, not compiled, not run on hardware
//
#include <hip/hip_runtime.h>

// ---------------------------------------------------------------------------
// MHA forward for gfx1250 (MI455X): bf16 WMMA, flash softmax, ping-pong
// double-buffered GEMM pipelines, K/V tiles staged into LDS via the Tensor
// Data Mover (fallback: async-to-LDS, fallback: ld+ds_store).
// ---------------------------------------------------------------------------

typedef __attribute__((ext_vector_type(16))) __bf16 v16bf;
typedef __attribute__((ext_vector_type(8)))  __bf16 v8bf;
typedef __attribute__((ext_vector_type(8)))  float  v8f;
typedef __attribute__((ext_vector_type(4)))  int    v4i;
typedef __attribute__((ext_vector_type(4)))  unsigned int u32x4;
typedef __attribute__((ext_vector_type(8)))  int    i32x8;

#define L_SEQ 2048
#define D_EMB 1024
#define NHEAD 16
#define FDIM  64
#define BATCH 2
#define MTOT  (BATCH * L_SEQ)   // 4096 total rows

#if defined(__has_builtin)
#if __has_builtin(__builtin_amdgcn_global_load_async_to_lds_b128) && \
    __has_builtin(__builtin_amdgcn_s_wait_asynccnt)
#define USE_ASYNC_LDS 1
#endif
#if __has_builtin(__builtin_amdgcn_tensor_load_to_lds) && \
    __has_builtin(__builtin_amdgcn_s_wait_tensorcnt)
#define HAVE_TDM 1
#endif
#endif
#ifndef USE_ASYNC_LDS
#define USE_ASYNC_LDS 0
#endif
#ifndef HAVE_TDM
#define HAVE_TDM 0
#endif

// therock-10 headers ship amd_gfx1250_TDM.h and use the 6-arg builtin;
// ROCm 7.2 clang-22 declares the 5-arg form.
#if defined(__has_include)
#if __has_include(<hip/amd_detail/amd_gfx1250_TDM.h>)
#define TDM_ARITY6 1
#endif
#endif
#ifndef TDM_ARITY6
#define TDM_ARITY6 0
#endif

typedef __attribute__((address_space(1))) v4i* gptr_v4i;
typedef __attribute__((address_space(3))) v4i* lptr_v4i;
typedef __attribute__((address_space(3))) __bf16* lptr_bf;

union FragPack { v16bf v; v8bf h[2]; };

static __device__ __forceinline__ __bf16 f2bf(float f) {
  unsigned u = __builtin_bit_cast(unsigned, f);
  u = (u + 0x7FFFu + ((u >> 16) & 1u)) >> 16;   // round-to-nearest-even
  unsigned short s = (unsigned short)u;
  return __builtin_bit_cast(__bf16, s);
}

static __device__ __forceinline__ v16bf load_frag16(const __bf16* p0, const __bf16* p1) {
  FragPack f;
  f.h[0] = *(const v8bf*)p0;
  f.h[1] = *(const v8bf*)p1;
  return f.v;
}

#define WMMA_BF16(A, B, C) \
  __builtin_amdgcn_wmma_f32_16x16x32_bf16(false, (A), false, (B), (short)0, (C), false, false)

// ----------------- staging paths: TDM > async-to-LDS > ld/st ---------------
#if HAVE_TDM
// One TDM descriptor-driven 2D tile load: tile_d1 rows x tile_d0 elems (bf16),
// row stride d0_stride elems, global tile start gsrc -> LDS ldst (packed).
static __device__ __forceinline__ void tdm_load_2d(const __bf16* gsrc, __bf16* ldst,
                                                   unsigned tile_d0, unsigned tile_d1,
                                                   unsigned tensor_d0, unsigned tensor_d1,
                                                   unsigned d0_stride) {
  const unsigned lds_off = (unsigned)(unsigned long long)(lptr_bf)ldst;
  const unsigned long long ga = (unsigned long long)gsrc;
  u32x4 g0;
  g0[0] = 1u;                                                   // count=1 (valid user D#)
  g0[1] = lds_off;                                              // lds_addr (bytes)
  g0[2] = (unsigned)ga;                                         // global_addr[31:0]
  g0[3] = (unsigned)((ga >> 32) & 0x01FFFFFFu) | (2u << 30);    // addr[56:32] | type=2
  i32x8 g1;
  g1[0] = (int)(1u << 16);                                      // wg_mask=0, data_size=2B
  g1[1] = (int)((tensor_d0 & 0xFFFFu) << 16);                   // tensor_dim0[15:0]
  g1[2] = (int)((tensor_d0 >> 16) | ((tensor_d1 & 0xFFFFu) << 16)); // dim0 hi / dim1 lo
  g1[3] = (int)((tensor_d1 >> 16) | (tile_d0 << 16));           // dim1 hi | tile_dim0
  g1[4] = (int)(tile_d1 & 0xFFFFu);                             // tile_dim1 (tile_dim2=0)
  g1[5] = (int)d0_stride;                                       // tensor_dim0_stride[31:0]
  g1[6] = 0;                                                    // stride hi / dim1_stride lo
  g1[7] = 0;
#if TDM_ARITY6
  __builtin_amdgcn_tensor_load_to_lds(g0, g1, (v4i)0, (v4i)0, (i32x8)0, 0);
#else
  __builtin_amdgcn_tensor_load_to_lds(g0, g1, (v4i)0, (v4i)0, 0);
#endif
}
#endif

// per-thread 16B global -> LDS (async when available)
static __device__ __forceinline__ void stage16(const __bf16* g, __bf16* l) {
#if USE_ASYNC_LDS
  __builtin_amdgcn_global_load_async_to_lds_b128((gptr_v4i)g, (lptr_v4i)l, 0, 0);
#else
  *(uint4*)l = *(const uint4*)g;
#endif
}

// ------------------------------- fp32 -> bf16 -------------------------------
__global__ void cvt_bf16_kernel(const float* __restrict__ in, __bf16* __restrict__ out, int n8) {
  int i = blockIdx.x * blockDim.x + threadIdx.x;
  if (i < n8) {
    const int base = i * 8;
    float4 f0 = *(const float4*)(in + base);
    float4 f1 = *(const float4*)(in + base + 4);
    v8bf o;
    o[0] = f2bf(f0.x); o[1] = f2bf(f0.y); o[2] = f2bf(f0.z); o[3] = f2bf(f0.w);
    o[4] = f2bf(f1.x); o[5] = f2bf(f1.y); o[6] = f2bf(f1.z); o[7] = f2bf(f1.w);
    *(v8bf*)(out + base) = o;
  }
}

// ------------------------------- QKV GEMM ----------------------------------
__global__ __launch_bounds__(256)
void qkv_gemm_kernel(const __bf16* __restrict__ xb, const __bf16* __restrict__ wb,
                     const float* __restrict__ bias,
                     __bf16* __restrict__ qo, __bf16* __restrict__ ko, __bf16* __restrict__ vo) {
  const int lane = threadIdx.x & 31;
  const int wave = threadIdx.x >> 5;
  const int wm = wave >> 1, wn = wave & 1;
  const int m0 = blockIdx.y * 128 + wm * 32;
  const int n0 = blockIdx.x * 128 + wn * 64;
  const int lm   = lane & 15;
  const int kb8  = (lane >> 4) << 3;   // A-frag K sub-base: 0 or 8
  const int kc16 = (lane >> 4) << 4;   // K sub-base 0 or 16 (unused; B-frag also uses kb8)

  v8f acc0[4] = {}, acc1[4] = {};
  const __bf16* arow0 = xb + (size_t)(m0 + lm) * D_EMB;
  const __bf16* arow1 = arow0 + (size_t)16 * D_EMB;
  const __bf16* brow[4];
#pragma unroll
  for (int j = 0; j < 4; ++j)
    brow[j] = wb + (size_t)(n0 + j * 16 + lm) * D_EMB + kb8;

#define LOADA0(k) load_frag16(arow0 + (k) + kb8, arow0 + (k) + 16 + kb8)
#define LOADA1(k) load_frag16(arow1 + (k) + kb8, arow1 + (k) + 16 + kb8)
#define LOADB(j, k) load_frag16(brow[j] + (k), brow[j] + (k) + 16)

  v16bf a0A = LOADA0(0), a1A = LOADA1(0);
  v16bf bA[4];
#pragma unroll
  for (int j = 0; j < 4; ++j) bA[j] = LOADB(j, 0);

  for (int k0 = 0; k0 < D_EMB; k0 += 64) {
    v16bf a0B = LOADA0(k0 + 32), a1B = LOADA1(k0 + 32);
    v16bf bB[4];
#pragma unroll
    for (int j = 0; j < 4; ++j) bB[j] = LOADB(j, k0 + 32);
#pragma unroll
    for (int j = 0; j < 4; ++j) {
      acc0[j] = WMMA_BF16(a0A, bA[j], acc0[j]);
      acc1[j] = WMMA_BF16(a1A, bA[j], acc1[j]);
    }
    const int kn = (k0 + 64 < D_EMB) ? (k0 + 64) : k0;
    a0A = LOADA0(kn); a1A = LOADA1(kn);
#pragma unroll
    for (int j = 0; j < 4; ++j) bA[j] = LOADB(j, kn);
#pragma unroll
    for (int j = 0; j < 4; ++j) {
      acc0[j] = WMMA_BF16(a0B, bB[j], acc0[j]);
      acc1[j] = WMMA_BF16(a1B, bB[j], acc1[j]);
    }
  }
#undef LOADA0
#undef LOADA1
#undef LOADB

#pragma unroll
  for (int j = 0; j < 4; ++j) {
    const int n  = n0 + j * 16 + lm;
    const float bv = bias[n];
    const int sec = n >> 10;          // 0=Q 1=K 2=V
    const int nn  = n & 1023;
    const int h = nn >> 6, f = nn & 63;
#pragma unroll
    for (int t = 0; t < 2; ++t) {
      const v8f& acc = t ? acc1[j] : acc0[j];
#pragma unroll
      for (int i = 0; i < 8; ++i) {
        const int r  = m0 + t * 16 + ((lane >> 4) << 3) + i;
        const int bb = r >> 11;
        const int ll = r & 2047;
        const __bf16 val = f2bf(acc[i] + bv);
        const size_t bh = (size_t)(bb * NHEAD + h);
        if (sec == 0)      qo[(bh * L_SEQ + ll) * FDIM + f] = val;
        else if (sec == 1) ko[(bh * L_SEQ + ll) * FDIM + f] = val;
        else               vo[(bh * FDIM + f) * L_SEQ + ll] = val;
      }
    }
  }
}

// ------------------------------ Flash attention ----------------------------
// grid: (L/128, B*H), block: 256 (8 waves, 16 query rows each).
// K/V tiles double-buffered in LDS; staged by TDM (wave 0) when available.
__global__ __launch_bounds__(256)
void attn_kernel(const __bf16* __restrict__ qm, const __bf16* __restrict__ km,
                 const __bf16* __restrict__ vtm, __bf16* __restrict__ ym) {
  __shared__ __bf16 kbuf[2][32][64];   // [buf][key][f]        8 KB
  __shared__ __bf16 vbuf[2][64][32];   // [buf][f][key]        8 KB
  __shared__ __bf16 pl[8][16][32];     // per-wave P transpose 8 KB
  const int tid  = threadIdx.x;
  const int lane = tid & 31;
  const int wave = tid >> 5;
  const int bh = blockIdx.y;
  const int b = bh >> 4, h = bh & 15;
  const int q0 = blockIdx.x * 128 + wave * 16;
  const int lm   = lane & 15;
  const int kb8  = (lane >> 4) << 3;
  const int kc16 = (lane >> 4) << 4;

  const __bf16* qb = qm + (size_t)bh * L_SEQ * FDIM;
  const __bf16* kb = km + (size_t)bh * L_SEQ * FDIM;
  const __bf16* vb = vtm + (size_t)bh * FDIM * L_SEQ;

  const __bf16* qrow = qb + (size_t)(q0 + lm) * FDIM;
  const v16bf aq0 = load_frag16(qrow + kc16,      qrow + kc16 + 8);
  const v16bf aq1 = load_frag16(qrow + 32 + kc16, qrow + 32 + kc16 + 8);

  const int krow = tid >> 3, kcol = (tid & 7) * 8;   // cooperative fallback split
  const int vrow = tid >> 2, vcol = (tid & 3) * 8;
  auto fill = [&](int buf, int kt) {
#if HAVE_TDM
    if (wave == 0) {
      // K tile: 32 keys x 64 f, row stride 64; V^T tile: 64 f x 32 keys, row stride 2048
      tdm_load_2d(kb + (size_t)kt * FDIM, &kbuf[buf][0][0], 64, 32, 64, L_SEQ, 64);
      tdm_load_2d(vb + kt, &vbuf[buf][0][0], 32, 64, L_SEQ, FDIM, L_SEQ);
    }
#else
    stage16(kb + (size_t)(kt + krow) * FDIM + kcol, &kbuf[buf][krow][kcol]);
    stage16(vb + (size_t)vrow * L_SEQ + kt + vcol,  &vbuf[buf][vrow][vcol]);
#endif
  };
  auto fill_wait = [&]() {
#if HAVE_TDM
    if (wave == 0) __builtin_amdgcn_s_wait_tensorcnt(0);
#elif USE_ASYNC_LDS
    __builtin_amdgcn_s_wait_asynccnt(0);
#endif
  };

  v8f o0 = {}, o1 = {}, o2 = {}, o3 = {};
  float mrow[8], lrow[8];
#pragma unroll
  for (int i = 0; i < 8; ++i) { mrow[i] = -__builtin_inff(); lrow[i] = 0.f; }

  fill(0, 0);
  fill_wait();
  __syncthreads();

  for (int kt = 0; kt < L_SEQ; kt += 32) {
    const int cur = (kt >> 5) & 1, nxt = cur ^ 1;
    const int kn = (kt + 32 < L_SEQ) ? kt + 32 : kt;
    const int kp = (kt + 256 < L_SEQ) ? kt + 256 : kt;
    // L2 prefetch far ahead (gfx1250 global_prefetch_b8)
    __builtin_prefetch(kb + (size_t)(kp + krow) * FDIM + kcol, 0, 1);
    __builtin_prefetch(vb + (size_t)vrow * L_SEQ + kp + vcol, 0, 1);
    // stage next tile (DMA) while computing the current one
    fill(nxt, kn);

    // current-V fragments from LDS (issued early; wait hidden behind softmax)
    v16bf vc0 = load_frag16(&vbuf[cur][lm][kc16],      &vbuf[cur][lm][kc16 + 8]);
    v16bf vc1 = load_frag16(&vbuf[cur][16 + lm][kc16], &vbuf[cur][16 + lm][kc16 + 8]);
    v16bf vc2 = load_frag16(&vbuf[cur][32 + lm][kc16], &vbuf[cur][32 + lm][kc16 + 8]);
    v16bf vc3 = load_frag16(&vbuf[cur][48 + lm][kc16], &vbuf[cur][48 + lm][kc16 + 8]);

    // ---- S = Q @ K^T for two 16-key tiles (K fragments from LDS) ----
    v8f s0 = {}, s1 = {};
    {
      v16bf kf;
      kf = load_frag16(&kbuf[cur][lm][kc16],           &kbuf[cur][lm][kc16 + 8]);
      s0 = WMMA_BF16(aq0, kf, s0);
      kf = load_frag16(&kbuf[cur][lm][32 + kc16],      &kbuf[cur][lm][32 + kc16 + 8]);
      s0 = WMMA_BF16(aq1, kf, s0);
      kf = load_frag16(&kbuf[cur][16 + lm][kc16],      &kbuf[cur][16 + lm][kc16 + 8]);
      s1 = WMMA_BF16(aq0, kf, s1);
      kf = load_frag16(&kbuf[cur][16 + lm][32 + kc16], &kbuf[cur][16 + lm][32 + kc16 + 8]);
      s1 = WMMA_BF16(aq1, kf, s1);
    }

    // ---- online softmax (row = (lane>=16)*8 + i, col = lane&15) ----
    const float scale = 0.125f;   // 1/sqrt(64)
#pragma unroll
    for (int i = 0; i < 8; ++i) {
      float a0 = s0[i] * scale, a1 = s1[i] * scale;
      float rm = fmaxf(a0, a1);
      for (int msk = 1; msk < 16; msk <<= 1) rm = fmaxf(rm, __shfl_xor(rm, msk, 32));
      float nm = fmaxf(mrow[i], rm);
      float corr = __expf(mrow[i] - nm);
      mrow[i] = nm;
      float p0 = __expf(a0 - nm);
      float p1 = __expf(a1 - nm);
      float rs = p0 + p1;
      for (int msk = 1; msk < 16; msk <<= 1) rs += __shfl_xor(rs, msk, 32);
      lrow[i] = lrow[i] * corr + rs;
      o0[i] *= corr; o1[i] *= corr; o2[i] *= corr; o3[i] *= corr;
      const int r = ((lane >> 4) << 3) + i;
      pl[wave][r][lm]      = f2bf(p0);
      pl[wave][r][16 + lm] = f2bf(p1);
    }
    // ---- re-read P in A-fragment layout (cross-lane transpose via LDS) ----
    v16bf pa = load_frag16(&pl[wave][lm][kc16], &pl[wave][lm][kc16 + 8]);
    // ---- O += P @ V ----
    o0 = WMMA_BF16(pa, vc0, o0);
    o1 = WMMA_BF16(pa, vc1, o1);
    o2 = WMMA_BF16(pa, vc2, o2);
    o3 = WMMA_BF16(pa, vc3, o3);

    fill_wait();        // next tile fully in LDS (issuing wave's DMA done)
    __syncthreads();    // all waves done reading cur + tile visible to all
  }
  // ---- normalize and store y[b, l, h*64 + f] as bf16 ----
#pragma unroll
  for (int i = 0; i < 8; ++i) {
    const float inv = 1.f / lrow[i];
    const int r = q0 + ((lane >> 4) << 3) + i;
    const size_t off = ((size_t)b * L_SEQ + r) * D_EMB + h * FDIM + lm;
    ym[off]      = f2bf(o0[i] * inv);
    ym[off + 16] = f2bf(o1[i] * inv);
    ym[off + 32] = f2bf(o2[i] * inv);
    ym[off + 48] = f2bf(o3[i] * inv);
  }
}

// ------------------------------- Output proj -------------------------------
__global__ __launch_bounds__(256)
void proj_gemm_kernel(const __bf16* __restrict__ yb, const __bf16* __restrict__ wb,
                      const float* __restrict__ bias, float* __restrict__ out) {
  const int lane = threadIdx.x & 31;
  const int wave = threadIdx.x >> 5;
  const int wm = wave >> 1, wn = wave & 1;
  const int m0 = blockIdx.y * 128 + wm * 32;
  const int n0 = blockIdx.x * 128 + wn * 64;
  const int lm   = lane & 15;
  const int kb8  = (lane >> 4) << 3;
  const int kc16 = (lane >> 4) << 4;

  v8f acc0[4] = {}, acc1[4] = {};
  const __bf16* arow0 = yb + (size_t)(m0 + lm) * D_EMB;
  const __bf16* arow1 = arow0 + (size_t)16 * D_EMB;
  const __bf16* brow[4];
#pragma unroll
  for (int j = 0; j < 4; ++j)
    brow[j] = wb + (size_t)(n0 + j * 16 + lm) * D_EMB + kb8;

#define LOADA0(k) load_frag16(arow0 + (k) + kb8, arow0 + (k) + 16 + kb8)
#define LOADA1(k) load_frag16(arow1 + (k) + kb8, arow1 + (k) + 16 + kb8)
#define LOADB(j, k) load_frag16(brow[j] + (k), brow[j] + (k) + 16)

  v16bf a0A = LOADA0(0), a1A = LOADA1(0);
  v16bf bA[4];
#pragma unroll
  for (int j = 0; j < 4; ++j) bA[j] = LOADB(j, 0);

  for (int k0 = 0; k0 < D_EMB; k0 += 64) {
    v16bf a0B = LOADA0(k0 + 32), a1B = LOADA1(k0 + 32);
    v16bf bB[4];
#pragma unroll
    for (int j = 0; j < 4; ++j) bB[j] = LOADB(j, k0 + 32);
#pragma unroll
    for (int j = 0; j < 4; ++j) {
      acc0[j] = WMMA_BF16(a0A, bA[j], acc0[j]);
      acc1[j] = WMMA_BF16(a1A, bA[j], acc1[j]);
    }
    const int kn = (k0 + 64 < D_EMB) ? (k0 + 64) : k0;
    a0A = LOADA0(kn); a1A = LOADA1(kn);
#pragma unroll
    for (int j = 0; j < 4; ++j) bA[j] = LOADB(j, kn);
#pragma unroll
    for (int j = 0; j < 4; ++j) {
      acc0[j] = WMMA_BF16(a0B, bB[j], acc0[j]);
      acc1[j] = WMMA_BF16(a1B, bB[j], acc1[j]);
    }
  }
#undef LOADA0
#undef LOADA1
#undef LOADB

#pragma unroll
  for (int j = 0; j < 4; ++j) {
    const int n = n0 + j * 16 + lm;
    const float bv = bias[n];
#pragma unroll
    for (int t = 0; t < 2; ++t) {
      const v8f& acc = t ? acc1[j] : acc0[j];
#pragma unroll
      for (int i = 0; i < 8; ++i) {
        const int r = m0 + t * 16 + ((lane >> 4) << 3) + i;
        out[(size_t)r * D_EMB + n] = acc[i] + bv;
      }
    }
  }
}

// --------------------------------- launch ----------------------------------
extern "C" void kernel_launch(void* const* d_in, const int* in_sizes, int n_in,
                              void* d_out, int out_size, void* d_ws, size_t ws_size,
                              hipStream_t stream) {
  const float* x      = (const float*)d_in[0];
  const float* w_qkv  = (const float*)d_in[1];
  const float* b_qkv  = (const float*)d_in[2];
  const float* w_proj = (const float*)d_in[3];
  const float* b_proj = (const float*)d_in[4];
  float* out = (float*)d_out;

  __bf16* ws = (__bf16*)d_ws;
  const size_t XB    = (size_t)MTOT * D_EMB;                   // 4M
  const size_t WQKV  = (size_t)3 * D_EMB * D_EMB;              // 3M
  const size_t WPROJ = (size_t)D_EMB * D_EMB;                  // 1M
  const size_t QSZ   = (size_t)BATCH * NHEAD * L_SEQ * FDIM;   // 4M
  __bf16* xb     = ws;
  __bf16* wqkvb  = xb + XB;
  __bf16* wprojb = wqkvb + WQKV;
  __bf16* qw     = wprojb + WPROJ;
  __bf16* kw     = qw + QSZ;
  __bf16* vtw    = kw + QSZ;
  __bf16* yw     = vtw + QSZ;

  cvt_bf16_kernel<<<(int)(XB / 8 / 256), 256, 0, stream>>>(x, xb, (int)(XB / 8));
  cvt_bf16_kernel<<<(int)(WQKV / 8 / 256), 256, 0, stream>>>(w_qkv, wqkvb, (int)(WQKV / 8));
  cvt_bf16_kernel<<<(int)(WPROJ / 8 / 256), 256, 0, stream>>>(w_proj, wprojb, (int)(WPROJ / 8));

  dim3 g1(3 * D_EMB / 128, MTOT / 128);   // 24 x 32
  qkv_gemm_kernel<<<g1, 256, 0, stream>>>(xb, wqkvb, b_qkv, qw, kw, vtw);

  dim3 g2(L_SEQ / 128, BATCH * NHEAD);    // 16 x 32
  attn_kernel<<<g2, 256, 0, stream>>>(qw, kw, vtw, yw);

  dim3 g3(D_EMB / 128, MTOT / 128);       // 8 x 32
  proj_gemm_kernel<<<g3, 256, 0, stream>>>(yw, wprojb, b_proj, out);
}
